// HyperAttention_21277267984600
// MI455X (gfx1250) — compile-verified, matrix operand order checked by reading, not run on hardware
//
#include <hip/hip_runtime.h>
#include <hip/hip_bf16.h>
#include <math.h>

// ---------------------------------------------------------------------------
// HyperAttention for MI455X (gfx1250): LSH bucket sort + block-diagonal
// attention + sampled residual, GEMMs via v_wmma_f32_16x16x32_bf16 (wave32).
// Row-sums on the matrix pipe (P x ones WMMA); single end-of-pass row-max
// reduction; sampled-gather set prefetched (global_prefetch_b8) during the
// block-diagonal phase.
// ---------------------------------------------------------------------------

typedef __bf16 bf16;
typedef __attribute__((ext_vector_type(16))) __bf16 v16bf;
typedef __attribute__((ext_vector_type(8)))  __bf16 v8bf;
typedef __attribute__((ext_vector_type(8)))  float  v8f;

#define B_   2
#define H_   16
#define BH_  (B_ * H_)
#define N_   8192
#define D_   64
#define NBLK 32          // N / 256
#define BS   256         // key block size == query block size
#define SAMP 256         // sampled columns
#define NPROJ 7
#define KT_STRIDE 72     // padded row stride (bf16 elems) for 256x64 tiles
#define VT_STRIDE 264    // padded row stride (bf16 elems) for 64x256 tiles

union V16U { v16bf v; v8bf h[2]; };

// Fragment load (A or B; symmetric layout). lane<16 holds row (row0+lane),
// K chunks {k0..k0+7, k0+16..k0+23}; lane>=16 holds same row, chunks +8/+24.
__device__ __forceinline__ v16bf ld_frag(const bf16* __restrict__ base,
                                         int row0, int k0, int stride) {
  int lane = threadIdx.x & 31;
  int lr = lane & 15;
  int half = lane >> 4;
  const bf16* p = base + (size_t)(row0 + lr) * stride + k0 + (half ? 8 : 0);
  V16U r;
  r.h[0] = *(const v8bf*)(p);
  r.h[1] = *(const v8bf*)(p + 16);
  return r.v;
}

__device__ __forceinline__ v16bf ones_frag() {
  V16U r;
#pragma unroll
  for (int i = 0; i < 16; ++i) r.v[i] = (bf16)1.0f;
  return r.v;
}

__device__ __forceinline__ v8f wmma_bf16(v16bf a, v16bf b, v8f c) {
  return __builtin_amdgcn_wmma_f32_16x16x32_bf16(
      false, a, false, b, (short)0, c, false, false);
}

// max-reduction across the 16-lane half (row-wise over N of a C fragment)
__device__ __forceinline__ float red_max16(float x) {
  for (int off = 8; off; off >>= 1) x = fmaxf(x, __shfl_xor(x, off, 32));
  return x;
}

// ---------------------------------------------------------------------------
// Kernel 1: LSH hash for Q and K.  bucket = code ^ (code>>1)  (Gray perm)
// ---------------------------------------------------------------------------
__global__ void lsh_hash_kernel(const float* __restrict__ q,
                                const float* __restrict__ k,
                                const float* __restrict__ pd,
                                int* __restrict__ hq, int* __restrict__ hk) {
  __shared__ float spd[D_ * NPROJ];
  for (int i = threadIdx.x; i < D_ * NPROJ; i += blockDim.x) spd[i] = pd[i];
  __syncthreads();
  const int total = BH_ * N_;
  int gi = blockIdx.x * blockDim.x + threadIdx.x;
  if (gi >= 2 * total) return;
  const float* x = (gi < total) ? (q + (size_t)gi * D_)
                                : (k + (size_t)(gi - total) * D_);
  float acc[NPROJ] = {0.f, 0.f, 0.f, 0.f, 0.f, 0.f, 0.f};
  for (int d = 0; d < D_; ++d) {
    float xv = x[d];
#pragma unroll
    for (int r = 0; r < NPROJ; ++r) acc[r] += xv * spd[d * NPROJ + r];
  }
  int code = 0;
#pragma unroll
  for (int r = 0; r < NPROJ; ++r) code |= (acc[r] > 0.f) ? (1 << r) : 0;
  int bucket = code ^ (code >> 1);
  if (gi < total) hq[gi] = bucket; else hk[gi - total] = bucket;
}

// ---------------------------------------------------------------------------
// Kernel 2: stable counting sort by bucket, one workgroup per (b,h).
// 128 buckets, bucket-per-thread ordered scatter => stable.
// ---------------------------------------------------------------------------
__global__ void sort_kernel(const int* __restrict__ hash, int* __restrict__ idx) {
  __shared__ int sh[N_];      // 32KB
  __shared__ int cnt[128];
  __shared__ int off[128];
  int bh = blockIdx.x;
  int tid = threadIdx.x;      // blockDim = 128
  cnt[tid] = 0;
  __syncthreads();
  for (int i = tid; i < N_; i += 128) {
    int h = hash[bh * N_ + i];
    sh[i] = h;
    atomicAdd(&cnt[h], 1);
  }
  __syncthreads();
  if (tid == 0) {
    int acc = 0;
    for (int b = 0; b < 128; ++b) { off[b] = acc; acc += cnt[b]; }
  }
  __syncthreads();
  int myoff = off[tid];
  int* outp = idx + bh * N_;
  for (int i = 0; i < N_; ++i) {
    if (sh[i] == tid) outp[myoff++] = i;   // increasing i => stable
  }
}

// ---------------------------------------------------------------------------
// Kernel 3: gather sorted q_s, k_s (row-major bf16) and v_sT (d-major bf16)
// ---------------------------------------------------------------------------
__global__ void gather_kernel(const float* __restrict__ q,
                              const float* __restrict__ k,
                              const float* __restrict__ v,
                              const int* __restrict__ qidx,
                              const int* __restrict__ kidx,
                              bf16* __restrict__ qs, bf16* __restrict__ ks,
                              bf16* __restrict__ vT) {
  int j = blockIdx.x * blockDim.x + threadIdx.x;   // sorted row over BH*N
  if (j >= BH_ * N_) return;
  int bh = j / N_;
  int r  = j % N_;
  size_t base = (size_t)bh * N_ * D_;
  int qi = qidx[j];
  int ki = kidx[j];
  const float* qp = q + base + (size_t)qi * D_;
  const float* kp = k + base + (size_t)ki * D_;
  const float* vp = v + base + (size_t)ki * D_;
  bf16* qo = qs + (size_t)j * D_;
  bf16* ko = ks + (size_t)j * D_;
  bf16* vo = vT + (size_t)bh * D_ * N_ + r;
#pragma unroll
  for (int d = 0; d < D_; ++d) {
    qo[d] = (bf16)qp[d];
    ko[d] = (bf16)kp[d];
    vo[(size_t)d * N_] = (bf16)vp[d];
  }
}

// ---------------------------------------------------------------------------
// Kernel 4: fused block-diagonal + sampled-residual attention, combine,
// un-sort scatter.  One workgroup (16 waves) per (bh, 256-row block); each
// wave owns a 16-row m-tile.  All GEMMs: v_wmma_f32_16x16x32_bf16.
// ---------------------------------------------------------------------------
__global__ void __launch_bounds__(512)
attn_kernel(const bf16* __restrict__ q_s, const bf16* __restrict__ k_s,
            const bf16* __restrict__ v_sT, const int* __restrict__ q_idx,
            const int* __restrict__ sampled, float* __restrict__ out) {
  __shared__ __align__(16) bf16 ktile[BS * KT_STRIDE];   // K block / k_sub
  __shared__ __align__(16) bf16 vtile[D_ * VT_STRIDE];   // V^T block / v_subT
  __shared__ __align__(16) bf16 pbuf[16 * 16 * 32];      // per-wave P staging
  __shared__ int sidx[SAMP];
  __shared__ int inblk[SAMP];
  __shared__ float s_logC;

  const int bh  = blockIdx.x / NBLK;
  const int blk = blockIdx.x % NBLK;
  const int tid  = threadIdx.x;
  const int wave = tid >> 5;
  const int lane = tid & 31;
  const int lr   = lane & 15;
  const int half = lane >> 4;
  const int qbase = blk * BS;
  const float scale = 0.125f;   // D^-0.5

  const bf16* qg = q_s  + (size_t)bh * N_ * D_;
  const bf16* kg = k_s  + (size_t)bh * N_ * D_;
  const bf16* vg = v_sT + (size_t)bh * D_ * N_;
  const int*  sp = sampled + bh * SAMP;

  // ---- stage K tile (256x64) and V^T tile (64x256) ----
  for (int c = tid; c < BS * (D_ / 8); c += blockDim.x) {
    int r = c >> 3, k8 = (c & 7) * 8;
    *(v8bf*)&ktile[r * KT_STRIDE + k8] = *(const v8bf*)&kg[(size_t)(qbase + r) * D_ + k8];
  }
  for (int c = tid; c < D_ * (BS / 8); c += blockDim.x) {
    int d = c >> 5, n8 = (c & 31) * 8;
    *(v8bf*)&vtile[d * VT_STRIDE + n8] = *(const v8bf*)&vg[(size_t)d * N_ + qbase + n8];
  }
  // ---- sampled ids -> LDS, in-block mask, prefetch phase-2 gather set ----
  for (int i = tid; i < SAMP; i += blockDim.x) {
    int si = sp[i];
    sidx[i] = si;
    inblk[i] = ((si >> 8) == blk) ? 1 : 0;
    __builtin_prefetch(&kg[(size_t)si * D_], 0, 3);        // global_prefetch_b8
    __builtin_prefetch(&vg[(size_t)(tid & 63) * N_ + si], 0, 3);
  }
  __syncthreads();
  if (tid == 0) {
    int cnt = 0;
    for (int i = 0; i < SAMP; ++i) cnt += inblk[i];
    s_logC = __logf((float)(N_ - BS) / (float)(SAMP - cnt));
  }

  // ---- Q fragments in registers (reused by both phases) ----
  const int r0 = wave * 16;
  v16bf aq0 = ld_frag(qg + (size_t)qbase * D_, r0, 0, D_);
  v16bf aq1 = ld_frag(qg + (size_t)qbase * D_, r0, 32, D_);
  const v16bf vones = ones_frag();

  bf16* mypb = pbuf + wave * (16 * 32);

  // ======================= phase 1: block-diagonal =======================
  float mb[8];
#pragma unroll
  for (int v = 0; v < 8; ++v) mb[v] = -1e30f;

  for (int j = 0; j < 16; ++j) {              // pass 1: per-lane running max
    v16bf b0 = ld_frag(ktile, j * 16, 0, KT_STRIDE);
    v16bf b1 = ld_frag(ktile, j * 16, 32, KT_STRIDE);
    v8f s = {};
    s = wmma_bf16(aq0, b0, s);
    s = wmma_bf16(aq1, b1, s);
#pragma unroll
    for (int v = 0; v < 8; ++v) mb[v] = fmaxf(mb[v], s[v] * scale);
  }
#pragma unroll
  for (int v = 0; v < 8; ++v) mb[v] = red_max16(mb[v]);   // one reduction

  v8f ob[4] = {};                              // 16x64 output accumulator
  v8f lacc_b = {};                             // row-sum accumulator (P x 1)
  for (int kc = 0; kc < 8; ++kc) {             // pass 2: exp, P stage, P*V
#pragma unroll
    for (int u = 0; u < 2; ++u) {
      int j = kc * 2 + u;
      v16bf b0 = ld_frag(ktile, j * 16, 0, KT_STRIDE);
      v16bf b1 = ld_frag(ktile, j * 16, 32, KT_STRIDE);
      v8f s = {};
      s = wmma_bf16(aq0, b0, s);
      s = wmma_bf16(aq1, b1, s);
#pragma unroll
      for (int v = 0; v < 8; ++v) {
        float p = __expf(s[v] * scale - mb[v]);
        int M = v + half * 8;
        mypb[M * 32 + u * 16 + lr] = (bf16)p;  // C-layout -> row-major restage
      }
    }
    asm volatile("s_wait_dscnt 0" ::: "memory");  // wave-private LDS RAW
    v16bf ap = ld_frag(mypb, 0, 0, 32);
    lacc_b = wmma_bf16(ap, vones, lacc_b);        // row-sum on the matrix pipe
#pragma unroll
    for (int dt = 0; dt < 4; ++dt) {
      v16bf bv = ld_frag(vtile, dt * 16, kc * 32, VT_STRIDE);
      ob[dt] = wmma_bf16(ap, bv, ob[dt]);
    }
  }
  __syncthreads();   // everyone done with ktile/vtile before re-staging

  // ---- stage k_sub (gathered rows) and v_subT (gathered cols) ----
  for (int c = tid; c < SAMP * (D_ / 8); c += blockDim.x) {
    int s_ = c >> 3, k8 = (c & 7) * 8;
    *(v8bf*)&ktile[s_ * KT_STRIDE + k8] = *(const v8bf*)&kg[(size_t)sidx[s_] * D_ + k8];
  }
  for (int c = tid; c < D_ * SAMP; c += blockDim.x) {
    int d = c >> 8, s_ = c & 255;
    vtile[d * VT_STRIDE + s_] = vg[(size_t)d * N_ + sidx[s_]];
  }
  __syncthreads();

  // ======================= phase 2: sampled residual =====================
  float mr[8];
#pragma unroll
  for (int v = 0; v < 8; ++v) mr[v] = -1e30f;

  for (int j = 0; j < 16; ++j) {               // pass 1: masked running max
    v16bf b0 = ld_frag(ktile, j * 16, 0, KT_STRIDE);
    v16bf b1 = ld_frag(ktile, j * 16, 32, KT_STRIDE);
    v8f s = {};
    s = wmma_bf16(aq0, b0, s);
    s = wmma_bf16(aq1, b1, s);
    int msk = inblk[j * 16 + lr];
#pragma unroll
    for (int v = 0; v < 8; ++v) {
      float t = msk ? -1e30f : s[v] * scale;
      mr[v] = fmaxf(mr[v], t);
    }
  }
#pragma unroll
  for (int v = 0; v < 8; ++v) mr[v] = red_max16(mr[v]);   // one reduction

  v8f orr[4] = {};
  v8f lacc_r = {};
  for (int kc = 0; kc < 8; ++kc) {
#pragma unroll
    for (int u = 0; u < 2; ++u) {
      int j = kc * 2 + u;
      v16bf b0 = ld_frag(ktile, j * 16, 0, KT_STRIDE);
      v16bf b1 = ld_frag(ktile, j * 16, 32, KT_STRIDE);
      v8f s = {};
      s = wmma_bf16(aq0, b0, s);
      s = wmma_bf16(aq1, b1, s);
      int msk = inblk[j * 16 + lr];
#pragma unroll
      for (int v = 0; v < 8; ++v) {
        float p = msk ? 0.f : __expf(s[v] * scale - mr[v]);
        int M = v + half * 8;
        mypb[M * 32 + u * 16 + lr] = (bf16)p;
      }
    }
    asm volatile("s_wait_dscnt 0" ::: "memory");
    v16bf ap = ld_frag(mypb, 0, 0, 32);
    lacc_r = wmma_bf16(ap, vones, lacc_r);
#pragma unroll
    for (int dt = 0; dt < 4; ++dt) {
      v16bf bv = ld_frag(vtile, dt * 16, kc * 32, VT_STRIDE);
      orr[dt] = wmma_bf16(ap, bv, orr[dt]);
    }
  }

  // ================= combine (logaddexp) + un-sort scatter ================
  const float logC = s_logC;
  float* outp = out + (size_t)bh * N_ * D_;
  int orow[8];
#pragma unroll
  for (int v = 0; v < 8; ++v) {
    int M = v + half * 8;
    orow[v] = q_idx[bh * N_ + qbase + r0 + M];
  }
#pragma unroll
  for (int v = 0; v < 8; ++v) {
    float lb = lacc_b[v];                      // row sum (same in all lanes)
    float lrr = lacc_r[v];
    float lse_b = mb[v] + __logf(lb);
    float lse_r = mr[v] + __logf(lrr) + logC;
    float Mx = fmaxf(lse_b, lse_r);
    float lse = Mx + __logf(__expf(lse_b - Mx) + __expf(lse_r - Mx));
    float cb = __expf(lse_b - lse);
    float ib = cb / lb;
    float ir = (1.f - cb) / lrr;
#pragma unroll
    for (int dt = 0; dt < 4; ++dt) {
      outp[(size_t)orow[v] * D_ + dt * 16 + lr] = ib * ob[dt][v] + ir * orr[dt][v];
    }
  }
}

// ---------------------------------------------------------------------------
extern "C" void kernel_launch(void* const* d_in, const int* in_sizes, int n_in,
                              void* d_out, int out_size, void* d_ws, size_t ws_size,
                              hipStream_t stream) {
  const float* q   = (const float*)d_in[0];
  const float* k   = (const float*)d_in[1];
  const float* v   = (const float*)d_in[2];
  const float* pd  = (const float*)d_in[3];
  const int*   smp = (const int*)d_in[4];
  float* out = (float*)d_out;

  const int total = BH_ * N_;
  char* ws = (char*)d_ws;
  int* hq   = (int*)ws;  ws += (size_t)total * sizeof(int);
  int* hk   = (int*)ws;  ws += (size_t)total * sizeof(int);
  int* qidx = (int*)ws;  ws += (size_t)total * sizeof(int);
  int* kidx = (int*)ws;  ws += (size_t)total * sizeof(int);
  bf16* qs = (bf16*)ws;  ws += (size_t)total * D_ * sizeof(bf16);
  bf16* ks = (bf16*)ws;  ws += (size_t)total * D_ * sizeof(bf16);
  bf16* vT = (bf16*)ws;  ws += (size_t)total * D_ * sizeof(bf16);
  (void)in_sizes; (void)n_in; (void)out_size; (void)ws_size;

  lsh_hash_kernel<<<(2 * total + 255) / 256, 256, 0, stream>>>(q, k, pd, hq, hk);
  sort_kernel<<<BH_, 128, 0, stream>>>(hq, qidx);
  sort_kernel<<<BH_, 128, 0, stream>>>(hk, kidx);
  gather_kernel<<<(total + 255) / 256, 256, 0, stream>>>(q, k, v, qidx, kidx, qs, ks, vT);
  attn_kernel<<<BH_ * NBLK, 512, 0, stream>>>(qs, ks, vT, qidx, smp, out);
}
